// GoalEncModel_49606872269243
// MI455X (gfx1250) — matrix operand order repaired and checked by reading, not compiled, not run on hardware
//
#include <hip/hip_runtime.h>
#include <hip/hip_bf16.h>

typedef __attribute__((ext_vector_type(16))) _Float16 v16h;
typedef __attribute__((ext_vector_type(8)))  _Float16 v8h;
typedef __attribute__((ext_vector_type(8)))  float    v8f;
typedef __attribute__((ext_vector_type(4)))  unsigned int u32x4;
typedef __attribute__((ext_vector_type(4)))  int      i32x4;
typedef __attribute__((ext_vector_type(8)))  int      i32x8;

#define HD   768
#define BB   64
#define LL   512
#define H3   2304
#define NROWS (LL*BB + BB)   // 32832 = 513 * 64

#if defined(__has_builtin)
#if __has_builtin(__builtin_amdgcn_tensor_load_to_lds) && \
    __has_builtin(__builtin_amdgcn_s_wait_tensorcnt)
#define HAVE_TDM 1
#endif
#endif
#ifndef HAVE_TDM
#define HAVE_TDM 0
#endif

// ---------------- WMMA fragment helpers (CDNA5 wave32 layouts) ----------------

// A-matrix 16x32 f16, row-major source. Lanes 0-15: row m, K {0..7, 16..23};
// lanes 16-31: row m, K {8..15, 24..31}.  (ISA 7.12.2)
__device__ __forceinline__ v16h load_a_frag(const _Float16* A, int lda, int row0,
                                            int k0, int lane) {
  int m   = row0 + (lane & 15);
  int off = (lane & 16) ? 8 : 0;
  const _Float16* p = A + (size_t)m * lda + k0 + off;
  v8h lo = *(const v8h*)(p);
  v8h hi = *(const v8h*)(p + 16);
  v16h r;
#pragma unroll
  for (int i = 0; i < 8; ++i) { r[i] = lo[i]; r[i + 8] = hi[i]; }
  return r;
}

// Same fragment but from an LDS-staged [64][32] f16 tile (ds_load_b128 x2).
__device__ __forceinline__ v16h load_a_lds(const _Float16* Ab, int mi, int lane) {
  int m   = mi * 16 + (lane & 15);
  int off = (lane & 16) ? 8 : 0;
  const _Float16* p = Ab + m * 32 + off;
  v8h lo = *(const v8h*)(p);
  v8h hi = *(const v8h*)(p + 16);
  v16h r;
#pragma unroll
  for (int i = 0; i < 8; ++i) { r[i] = lo[i]; r[i + 8] = hi[i]; }
  return r;
}

// B-matrix 32x16 f16 where column n of B == row n0+n of row-major W (computing X@W^T).
__device__ __forceinline__ v16h load_b_frag(const _Float16* W, int ldw, int n0,
                                            int k0, int lane) {
  int n    = n0 + (lane & 15);
  int koff = k0 + ((lane & 16) ? 8 : 0);
  const _Float16* p = W + (size_t)n * ldw + koff;
  v8h lo = *(const v8h*)(p);
  v8h hi = *(const v8h*)(p + 16);
  v16h r;
#pragma unroll
  for (int i = 0; i < 8; ++i) { r[i] = lo[i]; r[i + 8] = hi[i]; }
  return r;
}

__device__ __forceinline__ v8f wmma_f16(v16h a, v16h b, v8f c) {
  return __builtin_amdgcn_wmma_f32_16x16x32_f16(false, a, false, b, (short)0, c,
                                                false, false);
}

#if HAVE_TDM
// Issue a TDM 2D tile load (tile_k x tile_rows f16 elems) global -> LDS.
// D# per CDNA5 ISA: group0 = {count/lds_addr/global_addr/type},
// group1 = {data_size, tensor dims, tile dims, dim0 stride}. Groups 2/3 zero (2D).
__device__ __forceinline__ void tdm_load_tile_2d(unsigned lds_off,
                                                 const _Float16* gsrc,
                                                 unsigned tile_k, unsigned tile_rows,
                                                 unsigned row_stride_elems,
                                                 unsigned rows_total) {
  unsigned long long ga = (unsigned long long)(uintptr_t)gsrc;
  u32x4 g0;
  g0[0] = 1u;                                                // count=1, user mode
  g0[1] = lds_off;                                           // lds_addr (bytes)
  g0[2] = (unsigned)(ga & 0xFFFFFFFFu);                      // global_addr[31:0]
  g0[3] = (unsigned)((ga >> 32) & 0x01FFFFFFu) | (2u << 30); // addr[56:32] | type=2
  i32x8 g1;
  g1[0] = (int)(1u << 16);                                   // data_size=1 (2 bytes)
  g1[1] = (int)((row_stride_elems & 0xFFFFu) << 16);         // tensor_dim0[15:0]
  g1[2] = (int)(((row_stride_elems >> 16) & 0xFFFFu) |
                ((rows_total & 0xFFFFu) << 16));             // td0[31:16]|td1[15:0]
  g1[3] = (int)(((rows_total >> 16) & 0xFFFFu) |
                ((tile_k & 0xFFFFu) << 16));                 // td1[31:16]|tile0
  g1[4] = (int)(tile_rows & 0xFFFFu);                        // tile1 | tile2=0
  g1[5] = (int)row_stride_elems;                             // dim0_stride[31:0]
  g1[6] = 0;                                                 // stride hi | d1s lo
  g1[7] = 0;                                                 // d1 stride hi
  i32x4 z4 = {};
#if __clang_major__ >= 23
  i32x8 z8 = {};
  __builtin_amdgcn_tensor_load_to_lds(g0, g1, z4, z4, z8, 0);
#else
  __builtin_amdgcn_tensor_load_to_lds(g0, g1, z4, z4, 0);
#endif
}
#endif

// ---------------- small utility kernels ----------------

__global__ void f32_to_f16_kernel(const float* __restrict__ src,
                                  _Float16* __restrict__ dst, int n) {
  int i = blockIdx.x * blockDim.x + threadIdx.x;
  if (i < n) dst[i] = (_Float16)src[i];
}

__global__ void embed_relu_kernel(const int* __restrict__ goal,
                                  const float* __restrict__ emb,
                                  _Float16* __restrict__ X) {
  int idx = blockIdx.x * blockDim.x + threadIdx.x;
  if (idx >= NROWS * HD) return;
  int r = idx / HD;
  int h = idx - r * HD;
  float v;
  if (r < LL * BB) {
    int l = r / BB, b = r - l * BB;
    int tok = goal[b * LL + l];
    v = fmaxf(emb[(size_t)tok * HD + h], 0.f);
  } else {
    v = emb[(size_t)1 * HD + h];   // end token id = 1, not relu'd
  }
  X[idx] = (_Float16)v;
}

__global__ void zero_h_kernel(float* __restrict__ h32, _Float16* __restrict__ h16) {
  int i = blockIdx.x * blockDim.x + threadIdx.x;
  h32[i] = 0.f;
  h16[i] = (_Float16)0.f;
}

// ---------------- Gi = X @ w_ih^T + b_ih  (all 513 steps at once) ----------------
// grid = (513, 18); block = 256 (8 waves). Block tile: 64 rows x 128 cols.
// A slab (64x32 f16, 4KB) staged to LDS via TDM, double-buffered; wave w owns
// column strip w*16 (4 row tiles); 24 K-iterations.
__global__ void __launch_bounds__(256)
gi_gemm_kernel(const _Float16* __restrict__ X, const _Float16* __restrict__ Wih,
               const float* __restrict__ b_ih, _Float16* __restrict__ Gi) {
  __shared__ _Float16 Abuf[2][64 * 32];   // 2 x 4KB double buffer
  int row0 = blockIdx.x * 64;
  int lane = threadIdx.x & 31;
  int wave = threadIdx.x >> 5;
  int n0   = blockIdx.y * 128 + wave * 16;
  const _Float16* Ablk = X + (size_t)row0 * HD;

#if HAVE_TDM
  unsigned lds0 = (unsigned)(uintptr_t)(&Abuf[0][0]);
  unsigned lds1 = (unsigned)(uintptr_t)(&Abuf[1][0]);
  if (wave == 0) {
    tdm_load_tile_2d(lds0, Ablk, 32, 64, HD, NROWS - row0);
    __builtin_amdgcn_s_wait_tensorcnt(0);
  }
#else
  {
    // cooperative copy of k-slab 0: 2048 halves, 8 per thread
    int r = threadIdx.x >> 2, c = (threadIdx.x & 3) * 8;
    *(v8h*)(&Abuf[0][r * 32 + c]) = *(const v8h*)(Ablk + (size_t)r * HD + c);
  }
#endif
  __syncthreads();

  v8f acc[4] = {};
  for (int kk = 0; kk < HD / 32; ++kk) {
    int cur = kk & 1;
    if (kk + 1 < HD / 32) {
      int knext = (kk + 1) * 32;
#if HAVE_TDM
      if (wave == 0)
        tdm_load_tile_2d(cur ? lds0 : lds1, Ablk + knext, 32, 64, HD, NROWS - row0);
#else
      int r = threadIdx.x >> 2, c = (threadIdx.x & 3) * 8;
      *(v8h*)(&Abuf[cur ^ 1][r * 32 + c]) =
          *(const v8h*)(Ablk + (size_t)r * HD + knext + c);
#endif
    }
    v16h bf = load_b_frag(Wih, HD, n0, kk * 32, lane);
#pragma unroll
    for (int mi = 0; mi < 4; ++mi) {
      v16h af = load_a_lds(&Abuf[cur][0], mi, lane);
      acc[mi] = wmma_f16(af, bf, acc[mi]);
    }
#if HAVE_TDM
    if (wave == 0) __builtin_amdgcn_s_wait_tensorcnt(0);
#endif
    __syncthreads();
  }

  int col    = n0 + (lane & 15);
  float bias = b_ih[col];
  int rbase  = (lane & 16) ? 8 : 0;
#pragma unroll
  for (int mi = 0; mi < 4; ++mi)
#pragma unroll
    for (int v = 0; v < 8; ++v)
      Gi[(size_t)(row0 + mi * 16 + rbase + v) * H3 + col] =
          (_Float16)(acc[mi][v] + bias);
}

// ---------------- one GRU step (fused gh-GEMM + gates) ----------------
// grid = 12 (j-slices of 64); block = 256 (8 waves).
__global__ void __launch_bounds__(256)
gru_step_kernel(const _Float16* __restrict__ h16, const float* __restrict__ h32,
                const _Float16* __restrict__ Whh, const float* __restrict__ b_hh,
                const _Float16* __restrict__ Gi_step,  // [64, 3H] for this step
                float* __restrict__ hn32, _Float16* __restrict__ hn16,
                int apply_relu) {
  __shared__ float gh[64 * 192];   // 48 KB: r|z|n pre-activations for this j-slice
  int j0    = blockIdx.x * 64;
  int lane  = threadIdx.x & 31;
  int wave  = threadIdx.x >> 5;
  int mi    = wave >> 1;
  int nbase = (wave & 1) * 96;

  v8f acc[6] = {};
  for (int k0 = 0; k0 < HD; k0 += 32) {
    v16h af = load_a_frag(h16, HD, mi * 16, k0, lane);
#pragma unroll
    for (int t = 0; t < 6; ++t) {
      int n     = nbase + t * 16;          // 0..191
      int gate  = n >> 6;                  // tile never straddles a gate boundary
      int wrow0 = gate * HD + j0 + (n & 63);
      v16h bf = load_b_frag(Whh, HD, wrow0, k0, lane);
      acc[t] = wmma_f16(af, bf, acc[t]);
    }
  }

  int rbase = mi * 16 + ((lane & 16) ? 8 : 0);
#pragma unroll
  for (int t = 0; t < 6; ++t) {
    int n    = nbase + t * 16;
    int col  = n + (lane & 15);
    int gate = n >> 6;
    float bias = b_hh[gate * HD + j0 + (n & 63) + (lane & 15)];
#pragma unroll
    for (int v = 0; v < 8; ++v)
      gh[(rbase + v) * 192 + col] = acc[t][v] + bias;
  }
  __syncthreads();

  for (int idx = threadIdx.x; idx < 64 * 64; idx += 256) {
    int b = idx >> 6, j = idx & 63;
    float gr = gh[b * 192 + j];
    float gz = gh[b * 192 + 64 + j];
    float gn = gh[b * 192 + 128 + j];
    const _Float16* gi = Gi_step + (size_t)b * H3;
    float ir  = (float)gi[j0 + j];
    float iz  = (float)gi[HD + j0 + j];
    float in_ = (float)gi[2 * HD + j0 + j];
    float r  = 1.f / (1.f + __expf(-(ir + gr)));
    float z  = 1.f / (1.f + __expf(-(iz + gz)));
    float nn = tanhf(in_ + r * gn);
    float hv = (1.f - z) * nn + z * h32[b * HD + j0 + j];
    if (apply_relu) hv = fmaxf(hv, 0.f);
    hn32[b * HD + j0 + j] = hv;
    hn16[b * HD + j0 + j] = (_Float16)hv;
  }
}

// ---------------- scorer layers: Y = relu(X @ W^T + b), M=64, K=768 ----------------
__global__ void __launch_bounds__(256)
mlp_gemm_relu_kernel(const _Float16* __restrict__ Xin, const _Float16* __restrict__ W,
                     const float* __restrict__ bias, _Float16* __restrict__ Yout) {
  int lane = threadIdx.x & 31;
  int n0   = blockIdx.x * 128 + (threadIdx.x >> 5) * 16;
  v8f acc[4] = {};
  for (int k0 = 0; k0 < HD; k0 += 32) {
    v16h bf = load_b_frag(W, HD, n0, k0, lane);
#pragma unroll
    for (int mi = 0; mi < 4; ++mi) {
      v16h af = load_a_frag(Xin, HD, mi * 16, k0, lane);
      acc[mi] = wmma_f16(af, bf, acc[mi]);
    }
  }
  int col   = n0 + (lane & 15);
  float bv  = bias[col];
  int rbase = (lane & 16) ? 8 : 0;
#pragma unroll
  for (int mi = 0; mi < 4; ++mi)
#pragma unroll
    for (int v = 0; v < 8; ++v)
      Yout[(size_t)(mi * 16 + rbase + v) * HD + col] =
          (_Float16)fmaxf(acc[mi][v] + bv, 0.f);
}

// ---------------- final linear: scores[b] = y1[b,:] . swo + sbo ----------------
__global__ void final_dot_kernel(const _Float16* __restrict__ Y,
                                 const float* __restrict__ swo,
                                 const float* __restrict__ sbo,
                                 float* __restrict__ out) {
  int lane = threadIdx.x & 31;
  int wave = threadIdx.x >> 5;
  for (int b = wave; b < BB; b += 8) {
    float s = 0.f;
    for (int k = lane; k < HD; k += 32) s += (float)Y[b * HD + k] * swo[k];
#pragma unroll
    for (int off = 16; off > 0; off >>= 1) s += __shfl_down(s, off, 32);
    if (lane == 0) out[b] = s + sbo[0];
  }
}

// ---------------- host launch ----------------

extern "C" void kernel_launch(void* const* d_in, const int* in_sizes, int n_in,
                              void* d_out, int out_size, void* d_ws, size_t ws_size,
                              hipStream_t stream) {
  const int*   goal = (const int*)d_in[0];
  const float* emb  = (const float*)d_in[1];
  const float* w_ih = (const float*)d_in[2];
  const float* w_hh = (const float*)d_in[3];
  const float* b_ih = (const float*)d_in[4];
  const float* b_hh = (const float*)d_in[5];
  const float* sw0  = (const float*)d_in[6];
  const float* sb0  = (const float*)d_in[7];
  const float* sw1  = (const float*)d_in[8];
  const float* sb1  = (const float*)d_in[9];
  const float* swo  = (const float*)d_in[10];
  const float* sbo  = (const float*)d_in[11];
  (void)in_sizes; (void)n_in; (void)out_size; (void)ws_size;

  char* ws = (char*)d_ws;
  size_t off = 0;
  auto alloc = [&](size_t bytes) -> void* {
    void* p = ws + off;
    off = (off + bytes + 255) & ~(size_t)255;
    return p;
  };
  _Float16* X    = (_Float16*)alloc((size_t)NROWS * HD * 2);  // ~50 MB
  _Float16* Wih  = (_Float16*)alloc((size_t)H3 * HD * 2);
  _Float16* Whh  = (_Float16*)alloc((size_t)H3 * HD * 2);
  _Float16* Sw0h = (_Float16*)alloc((size_t)HD * HD * 2);
  _Float16* Sw1h = (_Float16*)alloc((size_t)HD * HD * 2);
  _Float16* Gi   = (_Float16*)alloc((size_t)NROWS * H3 * 2);  // ~151 MB
  float*    hA32 = (float*)alloc((size_t)BB * HD * 4);
  float*    hB32 = (float*)alloc((size_t)BB * HD * 4);
  _Float16* hA16 = (_Float16*)alloc((size_t)BB * HD * 2);
  _Float16* hB16 = (_Float16*)alloc((size_t)BB * HD * 2);
  _Float16* y0   = (_Float16*)alloc((size_t)BB * HD * 2);
  _Float16* y1   = (_Float16*)alloc((size_t)BB * HD * 2);

  int nW = H3 * HD;
  f32_to_f16_kernel<<<(nW + 255) / 256, 256, 0, stream>>>(w_ih, Wih, nW);
  f32_to_f16_kernel<<<(nW + 255) / 256, 256, 0, stream>>>(w_hh, Whh, nW);
  int nS = HD * HD;
  f32_to_f16_kernel<<<(nS + 255) / 256, 256, 0, stream>>>(sw0, Sw0h, nS);
  f32_to_f16_kernel<<<(nS + 255) / 256, 256, 0, stream>>>(sw1, Sw1h, nS);

  int totX = NROWS * HD;
  embed_relu_kernel<<<(totX + 255) / 256, 256, 0, stream>>>(goal, emb, X);

  dim3 ggrid(NROWS / 64, H3 / 128);
  gi_gemm_kernel<<<ggrid, 256, 0, stream>>>(X, Wih, b_ih, Gi);

  zero_h_kernel<<<(BB * HD) / 256, 256, 0, stream>>>(hA32, hA16);

  float*    hi32 = hA32; _Float16* hi16 = hA16;
  float*    ho32 = hB32; _Float16* ho16 = hB16;
  for (int t = 0; t < LL; ++t) {
    gru_step_kernel<<<HD / 64, 256, 0, stream>>>(
        hi16, hi32, Whh, b_hh, Gi + (size_t)t * BB * H3, ho32, ho16, 1);
    float* t32 = hi32; hi32 = ho32; ho32 = t32;
    _Float16* t16 = hi16; hi16 = ho16; ho16 = t16;
  }
  gru_step_kernel<<<HD / 64, 256, 0, stream>>>(
      hi16, hi32, Whh, b_hh, Gi + (size_t)LL * BB * H3, ho32, ho16, 0);

  mlp_gemm_relu_kernel<<<HD / 128, 256, 0, stream>>>(ho16, Sw0h, sb0, y0);
  mlp_gemm_relu_kernel<<<HD / 128, 256, 0, stream>>>(y0, Sw1h, sb1, y1);
  final_dot_kernel<<<1, 256, 0, stream>>>(y1, swo, sbo, (float*)d_out);
}
